// POUVIN_45655502356509
// MI455X (gfx1250) — matrix operand order repaired and checked by reading, not compiled, not run on hardware
//
#include <hip/hip_runtime.h>
#include <hip/hip_bf16.h>
#include <math.h>

// ---------------------------------------------------------------------------
// CDNA5 WMMA helpers (gfx1250, wave32)
// ---------------------------------------------------------------------------
typedef __attribute__((ext_vector_type(16))) _Float16 v16h;
typedef __attribute__((ext_vector_type(8)))  _Float16 v8h;
typedef __attribute__((ext_vector_type(8)))  float    v8f;

__device__ __forceinline__ v8f wmma_f16(v16h a, v16h b, v8f c) {
    // D = A(16x32) * B(32x16) + C(16x16), f32 accumulate
    return __builtin_amdgcn_wmma_f32_16x16x32_f16(
        /*neg_a=*/false, a, /*neg_b=*/false, b,
        /*c_mod=*/(short)0, c, /*reuse_a=*/false, /*reuse_b=*/false);
}

// ---------------------------------------------------------------------------
// Kernel 1: fused conv3x3(CIN->150) [+relu] -> conv1x1(150->MOUT) [+epilogue]
// One block per batch element, 8 waves. All operands staged in LDS:
//   - input map zero-padded to 34x34 (f16)
//   - stage-1/2 weights pre-swizzled into WMMA A-fragment order (f16)
//   - hidden tile [N=16][K=160] so stage-2 B fragments are contiguous b128s
// ---------------------------------------------------------------------------
template<int CIN, int MOUT, bool RELU_MID, bool ZNORM>
__global__ __launch_bounds__(256)
void conv2_kernel(const float* __restrict__ xin, int in_batch_stride,
                  const float* __restrict__ W1,   // (150, CIN*9)
                  const float* __restrict__ B1,   // (150)
                  const float* __restrict__ W2,   // (MOUT, 150)
                  const float* __restrict__ B2,   // (MOUT)
                  float* __restrict__ out)        // ZNORM ? (B,1024,MOUT) : (B,MOUT,1024)
{
    constexpr int K1  = CIN * 9;
    constexpr int MT2 = (MOUT + 15) / 16;
    __shared__ alignas(32) _Float16 smap[CIN * 34 * 34];     // zero-padded input
    __shared__ alignas(32) _Float16 sA1[10 * 512];           // [m][lane][i] fragments
    __shared__ alignas(32) _Float16 sA2[MT2 * 5 * 512];      // [m2*5+k][lane][i]
    __shared__ float sB1[160];
    __shared__ float sB2[MT2 * 16];
    __shared__ alignas(32) _Float16 shid[8 * 16 * 160];      // per-wave [N=16][K=160]
    __shared__ float szbuf[ZNORM ? 8 * 32 * 16 : 1];         // per-wave stage-2 stage

    const int blk  = blockIdx.x;
    const int tid  = threadIdx.x;
    const int wave = tid >> 5;
    const int lane = tid & 31;
    const int lhalf = lane >> 4;   // 0 or 1
    const int lcol  = lane & 15;

    // ---- cooperative LDS setup (once per block, branchless clamped gathers) ----
    for (int idx = tid; idx < CIN * 1156; idx += 256) {
        int c = idx / 1156, r = idx - c * 1156;
        int y = r / 34 - 1, x = r - (y + 1) * 34 - 1;
        bool in = (y >= 0 && y < 32 && x >= 0 && x < 32);
        int yc = in ? y : 0, xc = in ? x : 0;
        float v = xin[(size_t)blk * in_batch_stride + c * 1024 + yc * 32 + xc];
        smap[idx] = (_Float16)(in ? v : 0.f);
    }
    for (int idx = tid; idx < 10 * 512; idx += 256) {
        int m = idx >> 9, l = (idx >> 4) & 31, i = idx & 15;
        int M = (l & 15) + m * 16;
        int K = i + 16 * (l >> 4);
        bool ok = (M < 150) && (K < K1);
        float w = W1[(ok ? M : 0) * K1 + (ok ? K : 0)];
        sA1[idx] = (_Float16)(ok ? w : 0.f);
    }
    for (int idx = tid; idx < MT2 * 5 * 512; idx += 256) {
        int f = idx >> 9, l = (idx >> 4) & 31, i = idx & 15;
        int m2 = f / 5, k = f - 5 * m2;
        int M = (l & 15) + m2 * 16;
        int K = k * 32 + i + 16 * (l >> 4);
        bool ok = (M < MOUT) && (K < 150);
        float w = W2[(ok ? M : 0) * 150 + (ok ? K : 0)];
        sA2[idx] = (_Float16)(ok ? w : 0.f);
    }
    for (int idx = tid; idx < 160; idx += 256) sB1[idx] = (idx < 150) ? B1[idx] : 0.f;
    for (int idx = tid; idx < MT2 * 16; idx += 256) sB2[idx] = (idx < MOUT) ? B2[idx] : 0.f;
    __syncthreads();

    _Float16* myhid = &shid[wave * 2560];

    for (int tile = wave; tile < 64; tile += 8) {
        const int pbase = tile * 16;
        const int px = pbase + lcol;
        const int y = px >> 5, x = px & 31;

        // ---- B fragment for stage-1 (im2col patch from padded LDS map) ----
        v16h bfrag;
#pragma unroll
        for (int i = 0; i < 16; ++i) {
            int K = 16 * lhalf + i;
            int Kc = (K < K1) ? K : 0;
            int c = Kc / 9, tap = Kc - 9 * c;
            _Float16 v = smap[c * 1156 + (y + tap / 3) * 34 + (x + tap % 3)];
            bfrag[i] = (K < K1) ? v : (_Float16)0.f;
        }

        // ---- stage 1: 10 WMMAs covering 150 hidden channels ----
#pragma unroll
        for (int m = 0; m < 10; ++m) {
            v16h afrag = *(const v16h*)&sA1[m * 512 + lane * 16];
            v8f h = {};
            h = wmma_f16(afrag, bfrag, h);
            v8h hh;
#pragma unroll
            for (int j = 0; j < 8; ++j) {
                int row = m * 16 + 8 * lhalf + j;     // padded rows carry exact 0
                float hv = h[j] + sB1[row];
                if (RELU_MID) hv = fmaxf(hv, 0.f);
                hh[j] = (_Float16)hv;
            }
            *(v8h*)&myhid[lcol * 160 + m * 16 + 8 * lhalf] = hh;   // 16B ds_store
        }
        __builtin_amdgcn_wave_barrier();   // LDS is in-order per wave; fence compiler

        // ---- stage 2: 1x1 conv GEMM over K=160 (5 WMMAs / M-tile) ----
#pragma unroll
        for (int m2 = 0; m2 < MT2; ++m2) {
            v8f acc = {};
#pragma unroll
            for (int k = 0; k < 5; ++k) {
                v16h a2 = *(const v16h*)&sA2[(m2 * 5 + k) * 512 + lane * 16];
                v16h b2 = *(const v16h*)&myhid[lcol * 160 + k * 32 + 16 * lhalf];
                acc = wmma_f16(a2, b2, acc);
            }
            if constexpr (!ZNORM) {
#pragma unroll
                for (int j = 0; j < 8; ++j) {
                    int row = m2 * 16 + 8 * lhalf + j;
                    if (row < MOUT)
                        out[((size_t)blk * MOUT + row) * 1024 + px] = acc[j] + sB2[row];
                }
            } else {
#pragma unroll
                for (int j = 0; j < 8; ++j) {
                    int row = m2 * 16 + 8 * lhalf + j;
                    szbuf[wave * 512 + row * 16 + lcol] =
                        (row < MOUT) ? acc[j] + sB2[row] : 0.f;
                }
            }
        }
        if constexpr (ZNORM) {
            __builtin_amdgcn_wave_barrier();
            if (lane < 16) {   // one pixel per lane: sigmoid + channel-normalize
                float vals[MOUT];
                float s = 0.f;
#pragma unroll
                for (int o = 0; o < MOUT; ++o) {
                    float z = 1.f / (1.f + expf(-szbuf[wave * 512 + o * 16 + lane]));
                    vals[o] = z; s += z;
                }
                float inv = 1.f / (s + 1e-8f);
                float* op = out + ((size_t)blk * 1024 + pbase + lane) * MOUT;
#pragma unroll
                for (int o = 0; o < MOUT; ++o) op[o] = vals[o] * inv;
            }
            __builtin_amdgcn_wave_barrier();
        }
    }
}

// ---------------------------------------------------------------------------
// Kernel 2: value iteration, fully resident in LDS.
// R staged via CDNA5 async load-to-LDS (ASYNCcnt). Padded 34x34 v-plane;
// each thread owns 4 consecutive pixels sharing an 18-tap neighbor window.
// ---------------------------------------------------------------------------
__global__ __launch_bounds__(256)
void vi_kernel(const float* __restrict__ R,     // (B,5,1024)
               const float* __restrict__ Tv_w,  // (5,9)
               const float* __restrict__ Tv_b,  // (5)
               const int*   __restrict__ vik,
               float* __restrict__ Qout)        // (B,1024,5)
{
    __shared__ float sR[5 * 1024];
    __shared__ float sv[34 * 34];
    const int b = blockIdx.x, tid = threadIdx.x;

    // async DMA of R into LDS (gfx1250 GLOBAL_LOAD_ASYNC_TO_LDS, ASYNCcnt)
    for (int idx = tid; idx < 5 * 1024; idx += 256) {
        unsigned lds_off = (unsigned)(unsigned long long)
            ((__attribute__((address_space(3))) float*)&sR[idx]);
        unsigned long long ga = (unsigned long long)(R + (size_t)b * 5120 + idx);
        asm volatile("global_load_async_to_lds_b32 %0, %1, off"
                     :: "v"(lds_off), "v"(ga) : "memory");
    }
    for (int idx = tid; idx < 34 * 34; idx += 256) sv[idx] = 0.f;

    float w[45], tb[5];
#pragma unroll
    for (int i = 0; i < 45; ++i) w[i] = Tv_w[i];
#pragma unroll
    for (int a = 0; a < 5; ++a) tb[a] = Tv_b[a];

    asm volatile("s_wait_asynccnt 0x0" ::: "memory");
    __syncthreads();

    const int K = vik[0];
    const int p0 = tid * 4;                 // 4 consecutive pixels, same row
    const int y0 = p0 >> 5, x0 = p0 & 31;
    float q[4][5];
#pragma unroll
    for (int pi = 0; pi < 4; ++pi)
#pragma unroll
        for (int a = 0; a < 5; ++a) q[pi][a] = 0.f;

    for (int it = 0; it < K; ++it) {
        float n[3][6];
#pragma unroll
        for (int r = 0; r < 3; ++r)
#pragma unroll
            for (int c = 0; c < 6; ++c)
                n[r][c] = sv[(y0 + r) * 34 + x0 + c];   // padded: no bounds checks
        float vn[4];
#pragma unroll
        for (int pi = 0; pi < 4; ++pi) {
            float vmax = -3.4e38f;
#pragma unroll
            for (int a = 0; a < 5; ++a) {
                float qa = tb[a] + sR[a * 1024 + p0 + pi];
#pragma unroll
                for (int tap = 0; tap < 9; ++tap)
                    qa = fmaf(w[a * 9 + tap], n[tap / 3][pi + tap % 3], qa);
                q[pi][a] = qa;
                vmax = fmaxf(vmax, qa);
            }
            vn[pi] = vmax;
        }
        __syncthreads();
#pragma unroll
        for (int pi = 0; pi < 4; ++pi) sv[(y0 + 1) * 34 + x0 + 1 + pi] = vn[pi];
        __syncthreads();
    }
#pragma unroll
    for (int pi = 0; pi < 4; ++pi)
#pragma unroll
        for (int a = 0; a < 5; ++a)
            Qout[((size_t)b * 1024 + p0 + pi) * 5 + a] = q[pi][a];
}

// ---------------------------------------------------------------------------
// Kernel 3: sequential belief filter (4 steps) + readout.
// ---------------------------------------------------------------------------
__device__ __forceinline__ float block_reduce(float v, float* red, int tid) {
    red[tid] = v; __syncthreads();
    for (int s = 128; s > 0; s >>= 1) {
        if (tid < s) red[tid] += red[tid + s];
        __syncthreads();
    }
    float r = red[0]; __syncthreads();
    return r;
}

__global__ __launch_bounds__(256)
void belief_kernel(const float* __restrict__ b0,    // (B,1024)
                   const int*   __restrict__ act,   // (B,S)
                   const float* __restrict__ obs,   // (B,S,4)
                   const int*   __restrict__ stepp,
                   const float* __restrict__ Tb_w, const float* __restrict__ Tb_b,
                   const float* __restrict__ O1_w, const float* __restrict__ O1_b,
                   const float* __restrict__ O3_w, const float* __restrict__ O3_b,
                   const float* __restrict__ FL_w, const float* __restrict__ FL_b,
                   const float* __restrict__ Z,     // (B,1024,17)
                   const float* __restrict__ Q,     // (B,1024,5)
                   float* __restrict__ out, int Btot)
{
    __shared__ float sb[34 * 34];            // zero-padded belief plane
    __shared__ float red[256];
    __shared__ float sh17[17], sw17[17], sqb[5];
    const int bb = blockIdx.x, tid = threadIdx.x;
    const int S = stepp[0];
    const int p0 = tid * 4;
    const int y0 = p0 >> 5, x0 = p0 & 31;

    for (int idx = tid; idx < 34 * 34; idx += 256) sb[idx] = 0.f;
    __syncthreads();
#pragma unroll
    for (int pi = 0; pi < 4; ++pi)
        sb[(y0 + 1) * 34 + x0 + 1 + pi] = b0[(size_t)bb * 1024 + p0 + pi];
    __syncthreads();

    for (int t = 0; t < S; ++t) {
        // --- w = softmax(tanh(o @ O1^T + b1) @ O3^T + b3) ---
        if (tid < 17) {
            const float* op = obs + ((size_t)bb * S + t) * 4;
            float h = O1_b[tid];
#pragma unroll
            for (int k = 0; k < 4; ++k) h = fmaf(O1_w[tid * 4 + k], op[k], h);
            sh17[tid] = tanhf(h);
        }
        __syncthreads();
        if (tid < 17) {
            float g = O3_b[tid];
#pragma unroll
            for (int j = 0; j < 17; ++j) g = fmaf(O3_w[tid * 17 + j], sh17[j], g);
            red[tid] = g;
        }
        __syncthreads();
        if (tid == 0) {
            float m = -3.4e38f;
            for (int i = 0; i < 17; ++i) m = fmaxf(m, red[i]);
            float s = 0.f;
            for (int i = 0; i < 17; ++i) { float e = expf(red[i] - m); sw17[i] = e; s += e; }
            float inv = 1.f / s;
            for (int i = 0; i < 17; ++i) sw17[i] *= inv;
        }
        __syncthreads();

        const int a_sel = act[(size_t)bb * S + t];

        // --- b_next = conv3x3(b)[a_sel] * (Z . w), then normalize ---
        float n[3][6];
#pragma unroll
        for (int r = 0; r < 3; ++r)
#pragma unroll
            for (int c = 0; c < 6; ++c) n[r][c] = sb[(y0 + r) * 34 + x0 + c];
        float bn[4]; float psum = 0.f;
#pragma unroll
        for (int pi = 0; pi < 4; ++pi) {
            float bp = Tb_b[a_sel];
#pragma unroll
            for (int tap = 0; tap < 9; ++tap)
                bp = fmaf(Tb_w[a_sel * 9 + tap], n[tap / 3][pi + tap % 3], bp);
            const float* zp = Z + ((size_t)bb * 1024 + p0 + pi) * 17;
            float zo = 0.f;
#pragma unroll
            for (int o = 0; o < 17; ++o) zo = fmaf(zp[o], sw17[o], zo);
            bn[pi] = bp * zo;
            psum += bn[pi];
        }
        float tot = block_reduce(psum, red, tid);
        float inv = 1.f / (tot + 1e-8f);
        __syncthreads();
#pragma unroll
        for (int pi = 0; pi < 4; ++pi) sb[(y0 + 1) * 34 + x0 + 1 + pi] = bn[pi] * inv;
        __syncthreads();

        // --- Qb[a] = sum_p Q[b,p,a] * b[p]; out = Qb @ FL^T + bias ---
        for (int a = 0; a < 5; ++a) {
            float ps = 0.f;
#pragma unroll
            for (int pi = 0; pi < 4; ++pi)
                ps = fmaf(Q[((size_t)bb * 1024 + p0 + pi) * 5 + a],
                          sb[(y0 + 1) * 34 + x0 + 1 + pi], ps);
            float r = block_reduce(ps, red, tid);
            if (tid == 0) sqb[a] = r;
            __syncthreads();
        }
        if (tid < 5) {
            float o = FL_b[tid];
#pragma unroll
            for (int qd = 0; qd < 5; ++qd) o = fmaf(FL_w[tid * 5 + qd], sqb[qd], o);
            out[((size_t)t * Btot + bb) * 5 + tid] = o;
        }
        __syncthreads();
    }
    // final belief appended after the (S*B,5) outputs
    const size_t boff = (size_t)S * Btot * 5;
#pragma unroll
    for (int pi = 0; pi < 4; ++pi)
        out[boff + (size_t)bb * 1024 + p0 + pi] = sb[(y0 + 1) * 34 + x0 + 1 + pi];
}

// ---------------------------------------------------------------------------
// Launch
// ---------------------------------------------------------------------------
extern "C" void kernel_launch(void* const* d_in, const int* in_sizes, int n_in,
                              void* d_out, int out_size, void* d_ws, size_t ws_size,
                              hipStream_t stream) {
    const float* map_in = (const float*)d_in[0];
    const float* b0     = (const float*)d_in[1];
    const int*   act    = (const int*)  d_in[2];
    const float* obs    = (const float*)d_in[3];
    const int*   stepS  = (const int*)  d_in[5];
    const int*   VI_k   = (const int*)  d_in[6];
    const float* Tb_w = (const float*)d_in[7],  *Tb_b = (const float*)d_in[8];
    const float* Tv_w = (const float*)d_in[9],  *Tv_b = (const float*)d_in[10];
    const float* Z1_w = (const float*)d_in[11], *Z1_b = (const float*)d_in[12];
    const float* Z2_w = (const float*)d_in[13], *Z2_b = (const float*)d_in[14];
    const float* O1_w = (const float*)d_in[15], *O1_b = (const float*)d_in[16];
    const float* O3_w = (const float*)d_in[17], *O3_b = (const float*)d_in[18];
    const float* R1_w = (const float*)d_in[19], *R1_b = (const float*)d_in[20];
    const float* R2_w = (const float*)d_in[21], *R2_b = (const float*)d_in[22];
    const float* FL_w = (const float*)d_in[23], *FL_b = (const float*)d_in[24];

    const int B = in_sizes[0] / 2048;   // 256

    float* Rws = (float*)d_ws;                       // (B,5,1024)
    float* Qws = Rws + (size_t)B * 5120;             // (B,1024,5)
    float* Zws = Qws + (size_t)B * 5120;             // (B,1024,17)

    // R = conv1x1(relu(conv3x3(map, 2->150)), 150->5)
    conv2_kernel<2, 5, true, false><<<B, 256, 0, stream>>>(
        map_in, 2048, R1_w, R1_b, R2_w, R2_b, Rws);
    // Z = channel-norm(sigmoid(conv1x1(conv3x3(map[:, :1], 1->150), 150->17)))
    conv2_kernel<1, 17, false, true><<<B, 256, 0, stream>>>(
        map_in, 2048, Z1_w, Z1_b, Z2_w, Z2_b, Zws);
    // 64-step value iteration in LDS
    vi_kernel<<<B, 256, 0, stream>>>(Rws, Tv_w, Tv_b, VI_k, Qws);
    // sequential belief filter + readout
    belief_kernel<<<B, 256, 0, stream>>>(
        b0, act, obs, stepS, Tb_w, Tb_b, O1_w, O1_b, O3_w, O3_b,
        FL_w, FL_b, Zws, Qws, (float*)d_out, B);
}
